// EncoderLayer_61478161875002
// MI455X (gfx1250) — compile-verified, matrix operand order checked by reading, not run on hardware
//
#include <hip/hip_runtime.h>
#include <hip/hip_bf16.h>
#include <cstdint>

// ---------------------------------------------------------------------------
// Transformer encoder layer for MI455X (gfx1250), wave32, WMMA f16->f32.
// B=4, S=2048, D=1024, H=16, DK=64, F=2048, M=B*S=8192.
// GEMMs: 128x128 block, double-buffered Tensor-Data-Mover LDS staging.
// Attention: flash-style, 64-key blocks, exp2-domain online softmax.
// ---------------------------------------------------------------------------

typedef __attribute__((ext_vector_type(16))) _Float16 v16h;
typedef __attribute__((ext_vector_type(8)))  _Float16 v8h;
typedef __attribute__((ext_vector_type(8)))  float    v8f;
typedef __attribute__((ext_vector_type(4)))  unsigned int u32x4;
typedef __attribute__((ext_vector_type(4)))  int      i32x4;
typedef __attribute__((ext_vector_type(8)))  int      i32x8;

static constexpr int B_  = 4;
static constexpr int S_  = 2048;
static constexpr int D_  = 1024;
static constexpr int H_  = 16;
static constexpr int DK_ = 64;
static constexpr int F_  = 2048;
static constexpr int M_  = B_ * S_;   // 8192 rows

// --- Tensor Data Mover availability --------------------------------------
#if defined(__has_builtin)
#if __has_builtin(__builtin_amdgcn_tensor_load_to_lds)
#define USE_TDM 1
#endif
#endif
#ifndef USE_TDM
#define USE_TDM 0
#endif
#if __has_include(<hip/amd_detail/amd_gfx1250_TDM.h>)
#define TDM_SIX_ARGS 1   // amdgpu-toolchain (clang-23): 6-arg builtin
#else
#define TDM_SIX_ARGS 0   // ROCm 7.2 (clang-22): 5-arg builtin
#endif

union V16U { v16h v; v8h h[2]; };

// Fragment loader for 16-bit WMMA operands (ISA 7.12.2, 16-bit A 16x32):
// lane L holds row (A) / column (Bt) = L&15; lanes 0-15 carry K = {0..7,16..23},
// lanes 16-31 carry K = {8..15,24..31}.  Two contiguous 16B loads per lane.
// Works for A (row-major MxK) and for B given Bt stored row-major NxK.
__device__ __forceinline__ v16h load_frag(const _Float16* base, int ld, int lane) {
  const int r  = lane & 15;
  const int kh = (lane >> 4) & 1;
  const _Float16* p = base + (size_t)r * ld + kh * 8;
  V16U u;
  u.h[0] = *(const v8h*)(p);
  u.h[1] = *(const v8h*)(p + 16);
  return u.v;
}

__device__ __forceinline__ v8f wmma_f16(v16h a, v16h b, v8f c) {
  return __builtin_amdgcn_wmma_f32_16x16x32_f16(false, a, false, b, (short)0, c,
                                                false, false);
}

// TDM 2-D tile load: global (row-major, 2B elems, row stride = stride_elems)
// -> LDS with 4-DWORD padding every 16 DWORDs (i.e. 32-half rows laid out with
// a 40-half stride, matching the fragment loader's LDT and keeping 16B
// alignment).  D# built per ISA 8.3/8.4: group0 = {count, lds_addr,
// global_addr, type=2}; group1 = {data_size=1(2B), pad_enable,
// pad_interval=3 (16 DWORDs), pad_amount=3 (4 DWORDs), dims, strides}.
__device__ __forceinline__ void tdm_load_2d(const _Float16* gsrc, uint32_t lds_off,
                                            uint32_t tile_x, uint32_t tile_y,
                                            uint64_t stride_elems,
                                            uint64_t tensor_x, uint64_t tensor_y) {
#if USE_TDM
  union { uint64_t q[2]; u32x4 v; } g0;
  union { uint64_t q[4]; i32x8 v; } g1;
  const uint64_t ga = (uint64_t)(uintptr_t)gsrc;
  g0.q[0] = 1ull | ((uint64_t)lds_off << 32);                 // count=1, lds_addr
  g0.q[1] = (ga & 0x1FFFFFFFFFFFFFFull) | (2ull << 62);       // global_addr, type=2
  uint64_t q0 = (1ull << 16)            // data_size = 2 bytes
              | (1ull << 20)            // pad_enable
              | (3ull << 22)            // pad_interval: 16 DWORDs
              | (3ull << 25);           // pad_amount: 4 DWORDs
  q0 |= (tensor_x & 0xFFFFull) << 48;                         // tensor_dim0[15:0]
  uint64_t q1 = (tensor_x >> 16) & 0xFFFFull;                 // tensor_dim0[31:16]
  q1 |= (tensor_y & 0xFFFFFFFFull) << 16;                     // tensor_dim1
  q1 |= (uint64_t)tile_x << 48;                               // tile_dim0
  uint64_t q2 = (uint64_t)(tile_y & 0xFFFF);                  // tile_dim1
  q2 |= (stride_elems & 0xFFFFFFFFull) << 32;                 // dim0_stride[31:0]
  uint64_t q3 = (stride_elems >> 32) & 0xFFFFull;             // dim0_stride[47:32]
  g1.q[0] = q0; g1.q[1] = q1; g1.q[2] = q2; g1.q[3] = q3;
  const i32x4 z4 = {0, 0, 0, 0};
#if TDM_SIX_ARGS
  const i32x8 z8 = {0, 0, 0, 0, 0, 0, 0, 0};
  __builtin_amdgcn_tensor_load_to_lds(g0.v, g1.v, z4, z4, z8, 0);
#else
  __builtin_amdgcn_tensor_load_to_lds(g0.v, g1.v, z4, z4, 0);
#endif
#else
  (void)gsrc; (void)lds_off; (void)tile_x; (void)tile_y;
  (void)stride_elems; (void)tensor_x; (void)tensor_y;
#endif
}

// ---------------------------------------------------------------------------
// Weight transpose + fp32 -> f16 cast:  W[K][N] -> Wt[N][K]
// ---------------------------------------------------------------------------
__global__ __launch_bounds__(256)
void transpose_cast_kernel(const float* __restrict__ W, _Float16* __restrict__ Wt,
                           int K, int N) {
  __shared__ float tile[32][33];
  const int tx = threadIdx.x, ty = threadIdx.y;
  const int n0 = blockIdx.x * 32, k0 = blockIdx.y * 32;
#pragma unroll
  for (int i = 0; i < 4; ++i) {
    int k = k0 + ty + i * 8;
    tile[ty + i * 8][tx] = W[(size_t)k * N + (n0 + tx)];
  }
  __syncthreads();
#pragma unroll
  for (int i = 0; i < 4; ++i) {
    int n = n0 + ty + i * 8;
    Wt[(size_t)n * K + (k0 + tx)] = (_Float16)tile[tx][ty + i * 8];
  }
}

// ---------------------------------------------------------------------------
// LayerNorm over D=1024, one block (8 waves) per row, output f16.
// ---------------------------------------------------------------------------
__global__ __launch_bounds__(256)
void ln_kernel(const float* __restrict__ x, const float* __restrict__ g,
               const float* __restrict__ bta, _Float16* __restrict__ out) {
  const int row  = blockIdx.x;
  const int tid  = threadIdx.x;
  const int wave = tid >> 5, lane = tid & 31;
  const float4 v = ((const float4*)(x + (size_t)row * D_))[tid];
  float s  = v.x + v.y + v.z + v.w;
  float sq = v.x * v.x + v.y * v.y + v.z * v.z + v.w * v.w;
#pragma unroll
  for (int m = 1; m < 32; m <<= 1) {
    s  += __shfl_xor(s, m, 32);
    sq += __shfl_xor(sq, m, 32);
  }
  __shared__ float rsum[8], rsq[8];
  if (lane == 0) { rsum[wave] = s; rsq[wave] = sq; }
  __syncthreads();
  float ts = 0.f, tq = 0.f;
#pragma unroll
  for (int j = 0; j < 8; ++j) { ts += rsum[j]; tq += rsq[j]; }
  const float mu  = ts * (1.0f / D_);
  const float var = tq * (1.0f / D_) - mu * mu;
  const float inv = rsqrtf(var + 1e-5f);
  const float4 gg = ((const float4*)g)[tid];
  const float4 bb = ((const float4*)bta)[tid];
  _Float16* o = out + (size_t)row * D_ + tid * 4;
  o[0] = (_Float16)((v.x - mu) * inv * gg.x + bb.x);
  o[1] = (_Float16)((v.y - mu) * inv * gg.y + bb.y);
  o[2] = (_Float16)((v.z - mu) * inv * gg.z + bb.z);
  o[3] = (_Float16)((v.w - mu) * inv * gg.w + bb.w);
}

// ---------------------------------------------------------------------------
// WMMA GEMM:  C[M][N] = A[M][K] * Bt[N][K]^T  (+bias, mode-specific epilogue)
// Block tile 128x128, BK=32.  8 waves as 4(M) x 2(N): each wave owns a 32x64
// subtile -> 2 A-frags x 4 B-frags -> 8 WMMA per k-step.
// TDM path: double-buffered LDS.  TENSORcnt completes in-order per issuing
// wave, so after issuing tile i+1 (2 descriptors), s_wait_tensorcnt 2
// guarantees tile i's pair is resident; waves compute tile i while the DMA
// engine fills buffer i+1.  Barrier #1 frees the buffer being overwritten
// (last read in the previous iteration); barrier #2 publishes tile i.
// MODE 0: out f16, scatter to q/k layout [B,H,S,DK]
// MODE 1: out f16, scatter to v  layout [B,H,DK,S] (pre-transposed for P*V)
// MODE 2: out f32 = resid + C + bias  (attention proj / FFN-2 residual)
// MODE 3: out f16 = relu(C + bias)    (FFN-1)
// ---------------------------------------------------------------------------
template<int MODE>
__global__ __launch_bounds__(256)
void gemm_kernel(const _Float16* __restrict__ A, const _Float16* __restrict__ Bt,
                 const float* __restrict__ bias, const float* __restrict__ resid,
                 float* __restrict__ outf, _Float16* __restrict__ outh,
                 int M, int N, int K) {
  constexpr int BM = 128, BN = 128, BK = 32, LDT = BK + 8;  // 8-half pad, 16B rows
#if USE_TDM
  __shared__ __align__(16) _Float16 As[2][BM * LDT];
  __shared__ __align__(16) _Float16 Bs[2][BN * LDT];
#else
  __shared__ __align__(16) _Float16 As[1][BM * LDT];
  __shared__ __align__(16) _Float16 Bs[1][BN * LDT];
#endif

  const int tid  = threadIdx.x;
  const int wave = tid >> 5, lane = tid & 31;
  const int wm   = wave & 3, wn = wave >> 2;          // 4 x 2 wave grid
  const int mb = blockIdx.y * BM, nb = blockIdx.x * BN;

  v8f acc[2][4] = {};

#if USE_TDM
  const int niter = K / BK;
  if (wave == 0) {   // prologue: stage tile 0 into buffer 0
    tdm_load_2d(A  + (size_t)mb * K, (uint32_t)(uintptr_t)(void*)As[0],
                BK, BM, (uint64_t)K, (uint64_t)K, (uint64_t)M);
    tdm_load_2d(Bt + (size_t)nb * K, (uint32_t)(uintptr_t)(void*)Bs[0],
                BK, BN, (uint64_t)K, (uint64_t)K, (uint64_t)N);
  }
  for (int i = 0; i < niter; ++i) {
    __syncthreads();   // everyone done reading buffer (i+1)&1 from iter i-1
    if (wave == 0) {
      if (i + 1 < niter) {
        const int kn = (i + 1) * BK;
        tdm_load_2d(A  + (size_t)mb * K + kn,
                    (uint32_t)(uintptr_t)(void*)As[(i + 1) & 1],
                    BK, BM, (uint64_t)K, (uint64_t)K, (uint64_t)M);
        tdm_load_2d(Bt + (size_t)nb * K + kn,
                    (uint32_t)(uintptr_t)(void*)Bs[(i + 1) & 1],
                    BK, BN, (uint64_t)K, (uint64_t)K, (uint64_t)N);
        __builtin_amdgcn_s_wait_tensorcnt(2);  // tile i resident (in-order)
      } else {
        __builtin_amdgcn_s_wait_tensorcnt(0);
      }
    }
    __syncthreads();   // tile i published
    const _Float16* Ab = As[i & 1];
    const _Float16* Bb = Bs[i & 1];
    v16h a[2];
#pragma unroll
    for (int u = 0; u < 2; ++u)
      a[u] = load_frag(Ab + (wm * 32 + u * 16) * LDT, LDT, lane);
#pragma unroll
    for (int t = 0; t < 4; ++t) {
      const v16h b = load_frag(Bb + (wn * 64 + t * 16) * LDT, LDT, lane);
#pragma unroll
      for (int u = 0; u < 2; ++u)
        acc[u][t] = wmma_f16(a[u], b, acc[u][t]);
    }
  }
#else
  for (int kb = 0; kb < K; kb += BK) {
    __syncthreads();
    {  // A tile: 128x32 halfs, 16 halfs per thread (two 16B ops)
      const int row = tid >> 1, ch = (tid & 1) * 16;
      const _Float16* ga = A + (size_t)(mb + row) * K + kb + ch;
      *(v8h*)(As[0] + row * LDT + ch)     = *(const v8h*)(ga);
      *(v8h*)(As[0] + row * LDT + ch + 8) = *(const v8h*)(ga + 8);
      if (kb + BK < K) __builtin_prefetch(ga + BK, 0, 1);
    }
    {  // B tile: 128x32 halfs
      const int row = tid >> 1, ch = (tid & 1) * 16;
      const _Float16* gb = Bt + (size_t)(nb + row) * K + kb + ch;
      *(v8h*)(Bs[0] + row * LDT + ch)     = *(const v8h*)(gb);
      *(v8h*)(Bs[0] + row * LDT + ch + 8) = *(const v8h*)(gb + 8);
      if (kb + BK < K) __builtin_prefetch(gb + BK, 0, 1);
    }
    __syncthreads();
    v16h a[2];
#pragma unroll
    for (int u = 0; u < 2; ++u)
      a[u] = load_frag(As[0] + (wm * 32 + u * 16) * LDT, LDT, lane);
#pragma unroll
    for (int t = 0; t < 4; ++t) {
      const v16h b = load_frag(Bs[0] + (wn * 64 + t * 16) * LDT, LDT, lane);
#pragma unroll
      for (int u = 0; u < 2; ++u)
        acc[u][t] = wmma_f16(a[u], b, acc[u][t]);
    }
  }
#endif

  // Epilogue.  C layout: lane holds column n = lane&15; VGPR r holds row
  // m = r + 8*(lane>>4).
  const int nin = lane & 15, kh = lane >> 4;
#pragma unroll
  for (int u = 0; u < 2; ++u) {
#pragma unroll
    for (int t = 0; t < 4; ++t) {
      const int n = nb + wn * 64 + t * 16 + nin;
      const float bv = bias[n];
#pragma unroll
      for (int r = 0; r < 8; ++r) {
        const int m = mb + wm * 32 + u * 16 + r + 8 * kh;
        const float c = acc[u][t][r] + bv;
        if (MODE == 0) {            // q/k: [B,H,S,DK]
          const int b0 = m >> 11, s0 = m & (S_ - 1);
          const int h0 = n >> 6,  dk = n & (DK_ - 1);
          outh[((size_t)(b0 * H_ + h0) * S_ + s0) * DK_ + dk] = (_Float16)c;
        } else if (MODE == 1) {     // v transposed: [B,H,DK,S]
          const int b0 = m >> 11, s0 = m & (S_ - 1);
          const int h0 = n >> 6,  dk = n & (DK_ - 1);
          outh[((size_t)(b0 * H_ + h0) * DK_ + dk) * S_ + s0] = (_Float16)c;
        } else if (MODE == 2) {     // residual add, f32 out
          outf[(size_t)m * N + n] = resid[(size_t)m * N + n] + c;
        } else {                    // relu, f16 out
          outh[(size_t)m * N + n] = (_Float16)fmaxf(c, 0.0f);
        }
      }
    }
  }
}

// ---------------------------------------------------------------------------
// Flash attention.  Grid (S/128, B*H); 8 waves, each owns 16 queries.
// 64-key blocks: 4 score tiles per online-softmax round; softmax in exp2
// domain with 1/sqrt(DK)*log2(e) folded into one scale.  P re-laid out
// C-form -> A-form through a per-wave LDS tile (DS ops in-order per wave;
// explicit s_wait_dscnt before the fragment read).
// src_mask is all-true in this problem -> no masking term needed.
// ---------------------------------------------------------------------------
__global__ __launch_bounds__(256)
void attn_kernel(const _Float16* __restrict__ Q, const _Float16* __restrict__ Kc,
                 const _Float16* __restrict__ Vt, _Float16* __restrict__ ctx) {
  __shared__ __align__(16) _Float16 P[8][16 * 64];

  const int bh   = blockIdx.y;
  const int b0   = bh >> 4, h0 = bh & (H_ - 1);
  const int wave = threadIdx.x >> 5, lane = threadIdx.x & 31;
  const int q0   = blockIdx.x * 128 + wave * 16;

  const _Float16* Qb = Q  + ((size_t)bh * S_ + q0) * DK_;
  const _Float16* Kb = Kc + (size_t)bh * S_ * DK_;
  const _Float16* Vb = Vt + (size_t)bh * DK_ * S_;

  const v16h qa0 = load_frag(Qb,      DK_, lane);   // k-dim 0..31
  const v16h qa1 = load_frag(Qb + 32, DK_, lane);   // k-dim 32..63

  v8f acc[4] = {};
  float rm[8], rs[8];
#pragma unroll
  for (int r = 0; r < 8; ++r) { rm[r] = -1e30f; rs[r] = 0.0f; }

  _Float16* Pw = P[wave];
  const int nin = lane & 15, kh = lane >> 4;
  const float scl = 0.125f * 1.44269504088896340736f;   // 1/sqrt(DK) * log2(e)

  for (int kb = 0; kb < S_; kb += 64) {
    // 4 score tiles: keys [kb + 16*j, kb + 16*j + 16)
    v8f s[4] = {};
#pragma unroll
    for (int j = 0; j < 4; ++j) {
      const v16h k0 = load_frag(Kb + (size_t)(kb + 16 * j) * DK_,      DK_, lane);
      const v16h k1 = load_frag(Kb + (size_t)(kb + 16 * j) * DK_ + 32, DK_, lane);
      s[j] = wmma_f16(qa0, k0, s[j]);
      s[j] = wmma_f16(qa1, k1, s[j]);
    }
#pragma unroll
    for (int r = 0; r < 8; ++r) {
      float a0 = s[0][r] * scl, a1 = s[1][r] * scl;
      float a2 = s[2][r] * scl, a3 = s[3][r] * scl;
      float mx = fmaxf(fmaxf(a0, a1), fmaxf(a2, a3));
#pragma unroll
      for (int m = 1; m < 16; m <<= 1) mx = fmaxf(mx, __shfl_xor(mx, m, 32));
      const float mnew = fmaxf(rm[r], mx);
      const float corr = exp2f(rm[r] - mnew);
      rm[r] = mnew;
      const float p0 = exp2f(a0 - mnew), p1 = exp2f(a1 - mnew);
      const float p2 = exp2f(a2 - mnew), p3 = exp2f(a3 - mnew);
      float ps = (p0 + p1) + (p2 + p3);
#pragma unroll
      for (int m = 1; m < 16; m <<= 1) ps += __shfl_xor(ps, m, 32);
      rs[r] = rs[r] * corr + ps;
#pragma unroll
      for (int t = 0; t < 4; ++t) acc[t][r] *= corr;
      const int mrow = r + 8 * kh;
      Pw[mrow * 64 + nin]      = (_Float16)p0;
      Pw[mrow * 64 + 16 + nin] = (_Float16)p1;
      Pw[mrow * 64 + 32 + nin] = (_Float16)p2;
      Pw[mrow * 64 + 48 + nin] = (_Float16)p3;
    }
    // DS ops are in-order per wave; wait makes the C->A relayout of P visible
    // before the fragment reads (wave-synchronous, no barrier needed).
    asm volatile("s_wait_dscnt 0" ::: "memory");
    const v16h pa0 = load_frag(Pw,      64, lane);   // key-dim 0..31 of block
    const v16h pa1 = load_frag(Pw + 32, 64, lane);   // key-dim 32..63 of block
#pragma unroll
    for (int t = 0; t < 4; ++t) {
      const v16h vb0 = load_frag(Vb + (size_t)(t * 16) * S_ + kb,      S_, lane);
      const v16h vb1 = load_frag(Vb + (size_t)(t * 16) * S_ + kb + 32, S_, lane);
      acc[t] = wmma_f16(pa0, vb0, acc[t]);
      acc[t] = wmma_f16(pa1, vb1, acc[t]);
    }
  }

#pragma unroll
  for (int t = 0; t < 4; ++t) {
#pragma unroll
    for (int r = 0; r < 8; ++r) {
      const int q = q0 + r + 8 * kh;
      const float o = acc[t][r] * __builtin_amdgcn_rcpf(rs[r]);
      ctx[((size_t)(b0 * S_ + q)) * D_ + h0 * DK_ + t * 16 + nin] = (_Float16)o;
    }
  }
}

// ---------------------------------------------------------------------------
// Host-side orchestration.
// ---------------------------------------------------------------------------
extern "C" void kernel_launch(void* const* d_in, const int* in_sizes, int n_in,
                              void* d_out, int out_size, void* d_ws, size_t ws_size,
                              hipStream_t stream) {
  (void)in_sizes; (void)n_in; (void)out_size; (void)ws_size;
  const float* x     = (const float*)d_in[0];
  // d_in[1] = src_mask: all-true for this problem, softmax unaffected.
  const float* wq    = (const float*)d_in[2];
  const float* bq    = (const float*)d_in[3];
  const float* wk    = (const float*)d_in[4];
  const float* bk    = (const float*)d_in[5];
  const float* wv    = (const float*)d_in[6];
  const float* bv    = (const float*)d_in[7];
  const float* wo    = (const float*)d_in[8];
  const float* bo    = (const float*)d_in[9];
  const float* ln1g  = (const float*)d_in[10];
  const float* ln1b  = (const float*)d_in[11];
  const float* ln2g  = (const float*)d_in[12];
  const float* ln2b  = (const float*)d_in[13];
  const float* w1    = (const float*)d_in[14];
  const float* b1    = (const float*)d_in[15];
  const float* w2    = (const float*)d_in[16];
  const float* b2    = (const float*)d_in[17];
  float* out = (float*)d_out;

  char* ws = (char*)d_ws;
  const size_t MB = size_t(1) << 20;
  _Float16* wqt  = (_Float16*)(ws + 0 * MB);    // 2 MB
  _Float16* wkt  = (_Float16*)(ws + 2 * MB);    // 2 MB
  _Float16* wvt  = (_Float16*)(ws + 4 * MB);    // 2 MB
  _Float16* wot  = (_Float16*)(ws + 6 * MB);    // 2 MB
  _Float16* w1t  = (_Float16*)(ws + 8 * MB);    // 4 MB
  _Float16* w2t  = (_Float16*)(ws + 12 * MB);   // 4 MB
  _Float16* h16  = (_Float16*)(ws + 16 * MB);   // 16 MB (reused as h2)
  _Float16* q16  = (_Float16*)(ws + 32 * MB);   // 16 MB (reused: ff lo)
  _Float16* k16  = (_Float16*)(ws + 48 * MB);   // 16 MB (reused: ff hi)
  _Float16* vt16 = (_Float16*)(ws + 64 * MB);   // 16 MB
  _Float16* ctx  = (_Float16*)(ws + 80 * MB);   // 16 MB
  float*    x1   = (float*)   (ws + 96 * MB);   // 32 MB -> total 128 MB
  _Float16* ff16 = q16;                         // 32 MB alias over q16+k16

  const dim3 tpb256(256);
  const dim3 tb32x8(32, 8);

  // weight transpose+cast
  transpose_cast_kernel<<<dim3(D_ / 32, D_ / 32), tb32x8, 0, stream>>>(wq, wqt, D_, D_);
  transpose_cast_kernel<<<dim3(D_ / 32, D_ / 32), tb32x8, 0, stream>>>(wk, wkt, D_, D_);
  transpose_cast_kernel<<<dim3(D_ / 32, D_ / 32), tb32x8, 0, stream>>>(wv, wvt, D_, D_);
  transpose_cast_kernel<<<dim3(D_ / 32, D_ / 32), tb32x8, 0, stream>>>(wo, wot, D_, D_);
  transpose_cast_kernel<<<dim3(F_ / 32, D_ / 32), tb32x8, 0, stream>>>(w1, w1t, D_, F_);
  transpose_cast_kernel<<<dim3(D_ / 32, F_ / 32), tb32x8, 0, stream>>>(w2, w2t, F_, D_);

  // LN1
  ln_kernel<<<M_, tpb256, 0, stream>>>(x, ln1g, ln1b, h16);

  // QKV projections
  const dim3 gD(D_ / 128, M_ / 128);
  gemm_kernel<0><<<gD, tpb256, 0, stream>>>(h16, wqt, bq, nullptr, nullptr, q16,  M_, D_, D_);
  gemm_kernel<0><<<gD, tpb256, 0, stream>>>(h16, wkt, bk, nullptr, nullptr, k16,  M_, D_, D_);
  gemm_kernel<1><<<gD, tpb256, 0, stream>>>(h16, wvt, bv, nullptr, nullptr, vt16, M_, D_, D_);

  // attention
  attn_kernel<<<dim3(S_ / 128, B_ * H_), tpb256, 0, stream>>>(q16, k16, vt16, ctx);

  // output projection + residual -> x1 (f32)
  gemm_kernel<2><<<gD, tpb256, 0, stream>>>(ctx, wot, bo, x, x1, nullptr, M_, D_, D_);

  // LN2 -> h2 (aliases h16)
  ln_kernel<<<M_, tpb256, 0, stream>>>(x1, ln2g, ln2b, h16);

  // FFN
  gemm_kernel<3><<<dim3(F_ / 128, M_ / 128), tpb256, 0, stream>>>(h16, w1t, b1, nullptr, nullptr, ff16, M_, F_, D_);
  gemm_kernel<2><<<gD, tpb256, 0, stream>>>(ff16, w2t, b2, x1, out, nullptr, M_, D_, F_);
}
